// SparseAttention_11098195493618
// MI455X (gfx1250) — matrix operand order repaired and checked by reading, not compiled, not run on hardware
//
#include <hip/hip_runtime.h>
#include <hip/hip_bf16.h>
#include <math.h>

#define N_NODES 20000
#define N_EDGES 320000
#define HEADS   8
#define DIM     64
#define TILE    16   // edges per block tile (N_EDGES % TILE == 0)

typedef __attribute__((ext_vector_type(16))) _Float16 v16h;
typedef __attribute__((ext_vector_type(4)))  _Float16 v4h;
typedef __attribute__((ext_vector_type(8)))  float    v8f;
typedef __attribute__((ext_vector_type(4)))  float    v4f;

// ---- monotone float <-> uint encoding for atomicMax on floats ----
__device__ __forceinline__ unsigned enc_f32(float f) {
    unsigned b = __float_as_uint(f);
    return b ^ ((unsigned)(((int)b) >> 31) | 0x80000000u);
}
__device__ __forceinline__ float dec_f32(unsigned u) {
    unsigned m = (unsigned)((~(int)u) >> 31) | 0x80000000u;
    return __uint_as_float(u ^ m);
}

// ---- block-wide staging of 16 gathered Q rows and K rows (f32->f16) per head,
//      vectorized: float4 global loads -> packed 4x f16 LDS stores ----
__device__ __forceinline__ void stage_tiles(
    const float* __restrict__ q, const float* __restrict__ k,
    const int* __restrict__ adj0, const int* __restrict__ adj1, int base,
    _Float16 (&qs)[HEADS][TILE][DIM], _Float16 (&ks)[HEADS][TILE][DIM],
    int (&src)[TILE], int (&dst)[TILE])
{
    const int tid = threadIdx.x;
    if (tid < TILE) { src[tid] = adj0[base + tid]; dst[tid] = adj1[base + tid]; }
    __syncthreads();
    // HEADS*TILE*(DIM/4) = 2048 float4 per array, 256 threads -> 8 iters
    #pragma unroll
    for (int it = 0; it < (HEADS * TILE * (DIM / 4)) / 256; ++it) {
        int idx = tid + it * 256;
        int d4 = idx & (DIM / 4 - 1);          // float4 index within 64-dim row
        int e  = (idx >> 4) & (TILE - 1);
        int h  = idx >> 8;
        v4f qf = *(const v4f*)&q[((long)src[e] * HEADS + h) * DIM + 4 * d4];
        v4f kf = *(const v4f*)&k[((long)dst[e] * HEADS + h) * DIM + 4 * d4];
        v4h qh = { (_Float16)qf.x, (_Float16)qf.y, (_Float16)qf.z, (_Float16)qf.w };
        v4h kh = { (_Float16)kf.x, (_Float16)kf.y, (_Float16)kf.z, (_Float16)kf.w };
        *(v4h*)&qs[h][e][4 * d4] = qh;
        *(v4h*)&ks[h][e][4 * d4] = kh;
    }
    __syncthreads();
}

// ---- one wave: 16x16x64 cross-products via 2x v_wmma_f32_16x16x32_f16,
//      return the diagonal element this lane owns (lanes 0-7: e=lane,
//      lanes 24-31: e=lane-16; other lanes return 0) ----
__device__ __forceinline__ float wmma_qk_diag(
    const _Float16 (&qrows)[TILE][DIM], const _Float16 (&krows)[TILE][DIM], int lane)
{
    const int m  = lane & 15;
    const int hi = lane >> 4;
    v8f c = {};
    #pragma unroll
    for (int kb = 0; kb < DIM; kb += 32) {
        v16h a, b;
        // A 16x32 f16 layout: lanes<16: v0-3 K=0..7, v4-7 K=16..23; lanes>=16: +8
        #pragma unroll
        for (int v = 0; v < 4; ++v) {
            a[2 * v]     = qrows[m][kb + hi * 8 + 2 * v];
            a[2 * v + 1] = qrows[m][kb + hi * 8 + 2 * v + 1];
            a[8 + 2 * v]     = qrows[m][kb + 16 + hi * 8 + 2 * v];
            a[8 + 2 * v + 1] = qrows[m][kb + 16 + hi * 8 + 2 * v + 1];
        }
        // B 32x16 f16 layout: column n = lane&15; same K order as A (v0-3 K=0..7, v4-7 K=16..23; lanes>=16: +8)
        #pragma unroll
        for (int t = 0; t < 16; ++t) b[t] = krows[m][kb + (t < 8 ? hi * 8 + t : 16 + hi * 8 + (t - 8))];
        c = __builtin_amdgcn_wmma_f32_16x16x32_f16(false, a, false, b,
                                                   (short)0, c, false, false);
    }
    // diagonal: C layout lane n(0-15) vgpr v -> (M=v, N=n); lanes 16-31 -> (M=v+8, N=n-16)
    const int want = (lane < 8) ? lane : ((lane >= 24) ? (lane - 24) : -1);
    float diag = 0.0f;
    #pragma unroll
    for (int i = 0; i < 8; ++i)
        if (want == i) diag = c[i];
    return diag * 0.125f;  // 1/sqrt(64)
}

__global__ void __launch_bounds__(256)
ga_init_kernel(float* __restrict__ out, unsigned* __restrict__ maxbuf,
               float* __restrict__ sumbuf, int out_n, int nh)
{
    int idx = blockIdx.x * 256 + threadIdx.x;
    if (idx < out_n) out[idx] = 0.0f;
    if (idx < nh) { maxbuf[idx] = 0u; sumbuf[idx] = 0.0f; }
}

__global__ void __launch_bounds__(256)
ga_qkmax_kernel(const float* __restrict__ q, const float* __restrict__ k,
                const int* __restrict__ adj0, const int* __restrict__ adj1,
                unsigned* __restrict__ maxbuf)
{
    __shared__ _Float16 qs[HEADS][TILE][DIM];
    __shared__ _Float16 ks[HEADS][TILE][DIM];
    __shared__ int src[TILE], dst[TILE];
    const int base = blockIdx.x * TILE;
    stage_tiles(q, k, adj0, adj1, base, qs, ks, src, dst);

    const int h    = threadIdx.x >> 5;   // wave -> head (wave32)
    const int lane = threadIdx.x & 31;
    const float qk = wmma_qk_diag(qs[h], ks[h], lane);

    const int e = (lane < 8) ? lane : ((lane >= 24) ? (lane - 16) : -1);
    if (e >= 0)
        atomicMax(&maxbuf[src[e] * HEADS + h], enc_f32(qk));
}

__global__ void __launch_bounds__(256)
ga_agg_kernel(const float* __restrict__ q, const float* __restrict__ k,
              const float* __restrict__ vv,
              const int* __restrict__ adj0, const int* __restrict__ adj1,
              const unsigned* __restrict__ maxbuf, float* __restrict__ sumbuf,
              float* __restrict__ out)
{
    __shared__ _Float16 qs[HEADS][TILE][DIM];
    __shared__ _Float16 ks[HEADS][TILE][DIM];
    __shared__ int src[TILE], dst[TILE];
    const int base = blockIdx.x * TILE;
    stage_tiles(q, k, adj0, adj1, base, qs, ks, src, dst);

    const int h    = threadIdx.x >> 5;
    const int lane = threadIdx.x & 31;
    const float qk_own = wmma_qk_diag(qs[h], ks[h], lane);

    #pragma unroll 1
    for (int e = 0; e < TILE; ++e) {
        const int srclane = (e < 8) ? e : (e + 16);
        const float qk = __shfl(qk_own, srclane, 32);
        const int i = src[e];
        const int j = dst[e];
        const float mval = dec_f32(maxbuf[i * HEADS + h]);
        const float ex = expf(qk - mval);
        if (lane == 0) atomicAdd(&sumbuf[i * HEADS + h], ex);
        const float* vrow = &vv[((long)j * HEADS + h) * DIM];
        float*       orow = &out[((long)i * HEADS + h) * DIM];
        atomicAdd(&orow[lane],      ex * vrow[lane]);
        atomicAdd(&orow[lane + 32], ex * vrow[lane + 32]);
    }
}

__global__ void __launch_bounds__(256)
ga_norm_kernel(float* __restrict__ out, const float* __restrict__ sumbuf, int out_n)
{
    int idx = blockIdx.x * 256 + threadIdx.x;
    if (idx < out_n) {
        float s = sumbuf[idx >> 6];                 // (node*HEADS + h)
        out[idx] = (s > 0.0f) ? out[idx] / s : 0.0f;
    }
}

extern "C" void kernel_launch(void* const* d_in, const int* in_sizes, int n_in,
                              void* d_out, int out_size, void* d_ws, size_t ws_size,
                              hipStream_t stream)
{
    const float* q  = (const float*)d_in[0];
    const float* k  = (const float*)d_in[1];
    const float* v  = (const float*)d_in[2];
    const int*  adj = (const int*)d_in[3];
    const int* adj0 = adj;
    const int* adj1 = adj + N_EDGES;
    float* out = (float*)d_out;

    unsigned* maxbuf = (unsigned*)d_ws;                       // N_NODES*HEADS u32
    float*    sumbuf = (float*)d_ws + (N_NODES * HEADS);      // N_NODES*HEADS f32

    const int nh    = N_NODES * HEADS;
    const int out_n = N_NODES * HEADS * DIM;

    ga_init_kernel<<<(out_n + 255) / 256, 256, 0, stream>>>(out, maxbuf, sumbuf, out_n, nh);

    const int ntiles = N_EDGES / TILE;                        // 20000, exact
    ga_qkmax_kernel<<<ntiles, 256, 0, stream>>>(q, k, adj0, adj1, maxbuf);
    ga_agg_kernel<<<ntiles, 256, 0, stream>>>(q, k, v, adj0, adj1, maxbuf, sumbuf, out);
    ga_norm_kernel<<<(out_n + 255) / 256, 256, 0, stream>>>(out, sumbuf, out_n);
}
